// Eagle3DraftAttention_12360915878730
// MI455X (gfx1250) — matrix operand order repaired and checked by reading, not compiled, not run on hardware
//
#include <hip/hip_runtime.h>
#include <hip/hip_bf16.h>
#include <math.h>

// ---------- types ----------
typedef __attribute__((ext_vector_type(16))) __bf16 v16bf;
typedef __attribute__((ext_vector_type(8)))  __bf16 v8bf;
typedef __attribute__((ext_vector_type(8)))  float  v8f;

__device__ __forceinline__ __bf16 f2bf(float f) {
  unsigned u = __builtin_bit_cast(unsigned, f);
  unsigned r = u + 0x7FFFu + ((u >> 16) & 1u);   // round-to-nearest-even
  unsigned short h = (unsigned short)(r >> 16);
  return __builtin_bit_cast(__bf16, h);
}

__device__ __forceinline__ v16bf cat8(v8bf lo, v8bf hi) {
  return __builtin_shufflevector(lo, hi, 0,1,2,3,4,5,6,7,8,9,10,11,12,13,14,15);
}

__device__ __forceinline__ v8f wmma_bf16(v16bf a, v16bf b, v8f c) {
  // D = A(16x32 bf16) x B(32x16 bf16) + C(16x16 f32)
  return __builtin_amdgcn_wmma_f32_16x16x32_bf16(false, a, false, b, (short)0, c, false, false);
}

// ---------- problem constants ----------
#define LSEQ 2048
#define CIN  4096     // 2*HIDDEN
#define HQ   32
#define HKV  8
#define DH   64
#define HD   (HQ*DH)  // 2048
#define KVD  (HKV*DH) // 512

// ---------- 1) fp32 -> bf16 cast ----------
__global__ void k_cast_bf16(const float* __restrict__ in, __bf16* __restrict__ out, int n) {
  int i = blockIdx.x * blockDim.x + threadIdx.x;
  if (i < n) out[i] = f2bf(in[i]);
}

// ---------- 2) C[M,N] = A[M,K](bf16,row) * B[N,K](bf16,row)^T ; C fp32 ----------
// one wave per 16x64 output panel: A fragment reused across 4 N-tiles (4 WMMA/k-step)
__global__ __launch_bounds__(256)
void k_gemm_bf16_nt(const __bf16* __restrict__ A, const __bf16* __restrict__ B,
                    float* __restrict__ C, int M, int N, int K) {
  const int wave = threadIdx.x >> 5;
  const int lane = threadIdx.x & 31;
  const int l16  = lane & 15;
  const int g    = lane >> 4;
  const int tilesN = N >> 6;                       // 64-wide panels
  const int tile = blockIdx.x * 8 + wave;
  if (tile >= (M >> 4) * tilesN) return;           // wave-uniform exit (EXEC all-1 for WMMA)
  const int row0 = (tile / tilesN) << 4;
  const int col0 = (tile % tilesN) << 6;
  const __bf16* arow = A + (size_t)(row0 + l16) * K;
  const __bf16* brow0 = B + (size_t)(col0 +  0 + l16) * K;
  const __bf16* brow1 = B + (size_t)(col0 + 16 + l16) * K;
  const __bf16* brow2 = B + (size_t)(col0 + 32 + l16) * K;
  const __bf16* brow3 = B + (size_t)(col0 + 48 + l16) * K;
  v8f acc0 = {}, acc1 = {}, acc2 = {}, acc3 = {};
  for (int k0 = 0; k0 < K; k0 += 32) {
    __builtin_prefetch(arow + k0 + 256, 0, 1);     // global_prefetch_b8
    __builtin_prefetch(brow0 + k0 + 256, 0, 1);
    // A fragment: lane-half g holds contiguous K run [16g,16g+16)
    v16bf a = cat8(*(const v8bf*)(arow + k0 + 16*g),
                   *(const v8bf*)(arow + k0 + 16*g + 8));
    // B fragments: lane-half g holds contiguous K run [16g,16g+16)
    v16bf b0 = cat8(*(const v8bf*)(brow0 + k0 + 16*g), *(const v8bf*)(brow0 + k0 + 16*g + 8));
    v16bf b1 = cat8(*(const v8bf*)(brow1 + k0 + 16*g), *(const v8bf*)(brow1 + k0 + 16*g + 8));
    v16bf b2 = cat8(*(const v8bf*)(brow2 + k0 + 16*g), *(const v8bf*)(brow2 + k0 + 16*g + 8));
    v16bf b3 = cat8(*(const v8bf*)(brow3 + k0 + 16*g), *(const v8bf*)(brow3 + k0 + 16*g + 8));
    acc0 = wmma_bf16(a, b0, acc0);
    acc1 = wmma_bf16(a, b1, acc1);
    acc2 = wmma_bf16(a, b2, acc2);
    acc3 = wmma_bf16(a, b3, acc3);
  }
#pragma unroll
  for (int r = 0; r < 8; ++r) {                    // C/D: row r+8g, col l16 (+16t)
    float* crow = C + (size_t)(row0 + r + 8*g) * N + col0 + l16;
    crow[0]  = acc0[r];
    crow[16] = acc1[r];
    crow[32] = acc2[r];
    crow[48] = acc3[r];
  }
}

// ---------- 3) RoPE + pack head-major bf16: out[h][l][d] ----------
__global__ void k_rope_pack(const float* __restrict__ Q, const float* __restrict__ cosb,
                            const float* __restrict__ sinb, __bf16* __restrict__ Qout,
                            int L, int nH) {
  size_t idx = (size_t)blockIdx.x * blockDim.x + threadIdx.x;
  size_t total = (size_t)L * nH * DH;
  if (idx >= total) return;
  int d = idx & (DH - 1);
  size_t t = idx >> 6;
  int h = (int)(t % nH);
  int l = (int)(t / nH);
  const float* qrow = Q + (size_t)l * (nH * DH) + h * DH;
  float c = cosb[l * DH + d];
  float s = sinb[l * DH + d];
  float q  = qrow[d];
  float qr = (d < DH/2) ? -qrow[d + DH/2] : qrow[d - DH/2];  // rotate_half
  Qout[((size_t)h * L + l) * DH + d] = f2bf(q * c + qr * s);
}

// ---------- 4) pack V transposed bf16: out[h][d][l] ----------
__global__ void k_vpack_t(const float* __restrict__ V, __bf16* __restrict__ Vt, int L) {
  size_t idx = (size_t)blockIdx.x * blockDim.x + threadIdx.x;
  size_t total = (size_t)L * KVD;
  if (idx >= total) return;
  int c = (int)(idx % KVD);
  int l = (int)(idx / KVD);
  int h = c >> 6, d = c & 63;
  Vt[((size_t)h * DH + d) * L + l] = f2bf(V[idx]);
}

// ---------- 5) causal flash attention, one wave per (head, 16-query tile) ----------
// V chunk is staged LDS-ward with async loads overlapped with the S=QK^T WMMAs.
__global__ __launch_bounds__(256)
void k_flash_attn(const __bf16* __restrict__ Qb,   // [HQ][L][64]
                  const __bf16* __restrict__ Kb,   // [HKV][L][64]
                  const __bf16* __restrict__ Vt,   // [HKV][64][L]
                  __bf16* __restrict__ Ob,         // [L][HQ*64]
                  int L) {
  __shared__ __align__(16) __bf16 Plds[8][16][32]; // per-wave P slab (16 q x 32 k)
  __shared__ __align__(16) __bf16 Vlds[8][64][32]; // per-wave V slab (64 d x 32 k)
  const int wave = threadIdx.x >> 5;
  const int lane = threadIdx.x & 31;
  const int l16 = lane & 15, g = lane >> 4;
  const int qtiles = L >> 4;
  const int gwave = blockIdx.x * 8 + wave;
  const int h  = gwave / qtiles;
  const int q0 = (gwave % qtiles) << 4;
  const int hkv = h >> 2;                          // n_rep = 4, kv-head-major repeat

  // preload Q fragments for K-dim 0..31 and 32..63
  const __bf16* qrow = Qb + ((size_t)h * L + q0 + l16) * DH;
  const v16bf aq0 = cat8(*(const v8bf*)(qrow + 16*g),      *(const v8bf*)(qrow + 16*g + 8));
  const v16bf aq1 = cat8(*(const v8bf*)(qrow + 32 + 16*g), *(const v8bf*)(qrow + 40 + 16*g));

  const __bf16* kbase = Kb + (size_t)hkv * L * DH;
  const __bf16* vbase = Vt + (size_t)hkv * DH * L;

  // per-lane async-copy coordinates: 64 rows x 4 segs(16B) ... unit = it*32+lane
  // row = unit >> 2, seg = unit & 3  (4 segs of 16B per 64B row)
  const unsigned vldsBase = (unsigned)(uintptr_t)&Vlds[wave][0][0];

  float mrow[8], lrow[8];
  v8f o[4] = {};
#pragma unroll
  for (int r = 0; r < 8; ++r) { mrow[r] = -INFINITY; lrow[r] = 0.f; }

  for (int kb = 0; kb < q0 + 16; kb += 32) {       // 32 keys per iteration
    // ---- kick off async V-chunk copy (global -> LDS), overlapped with S ----
#pragma unroll
    for (int it = 0; it < 8; ++it) {
      const int unit = it * 32 + lane;
      const int row = unit >> 2;                   // d row 0..63
      const int seg = unit & 3;                    // 16B segment in 64B row
      const __bf16* gsrc = vbase + (size_t)row * L + kb + seg * 8;
      const unsigned ldst = vldsBase + row * 64 + seg * 16;
      asm volatile("global_load_async_to_lds_b128 %0, %1, off"
                   :: "v"(ldst), "v"(gsrc) : "memory");
    }
    // ---- S = Q Kt : two 16-key tiles, K-dim 64 -> 2 WMMA each ----
    v8f s0 = {}, s1 = {};
    {
      const __bf16* kr0 = kbase + (size_t)(kb + l16) * DH;
      const __bf16* kr1 = kbase + (size_t)(kb + 16 + l16) * DH;
      v16bf b00 = cat8(*(const v8bf*)(kr0 + 16*g),      *(const v8bf*)(kr0 + 16*g + 8));
      v16bf b01 = cat8(*(const v8bf*)(kr0 + 32 + 16*g), *(const v8bf*)(kr0 + 40 + 16*g));
      s0 = wmma_bf16(aq0, b00, s0);
      s0 = wmma_bf16(aq1, b01, s0);
      v16bf b10 = cat8(*(const v8bf*)(kr1 + 16*g),      *(const v8bf*)(kr1 + 16*g + 8));
      v16bf b11 = cat8(*(const v8bf*)(kr1 + 32 + 16*g), *(const v8bf*)(kr1 + 40 + 16*g));
      s1 = wmma_bf16(aq0, b10, s1);
      s1 = wmma_bf16(aq1, b11, s1);
    }
    // ---- scale 1/sqrt(64), causal mask, online softmax ----
    const int key0 = kb + l16, key1 = kb + 16 + l16;
#pragma unroll
    for (int r = 0; r < 8; ++r) {
      const int q = q0 + r + 8*g;
      s0[r] = (key0 > q) ? -1e30f : s0[r] * 0.125f;
      s1[r] = (key1 > q) ? -1e30f : s1[r] * 0.125f;
      float mx = fmaxf(s0[r], s1[r]);              // row reduction over N (16 lanes/half)
      mx = fmaxf(mx, __shfl_xor(mx, 1, 32));
      mx = fmaxf(mx, __shfl_xor(mx, 2, 32));
      mx = fmaxf(mx, __shfl_xor(mx, 4, 32));
      mx = fmaxf(mx, __shfl_xor(mx, 8, 32));
      float mnew = fmaxf(mrow[r], mx);
      float corr = __expf(mrow[r] - mnew);
      mrow[r] = mnew;
      float p0 = __expf(s0[r] - mnew);
      float p1 = __expf(s1[r] - mnew);
      s0[r] = p0; s1[r] = p1;
      float ps = p0 + p1;
      ps += __shfl_xor(ps, 1, 32);
      ps += __shfl_xor(ps, 2, 32);
      ps += __shfl_xor(ps, 4, 32);
      ps += __shfl_xor(ps, 8, 32);
      lrow[r] = lrow[r] * corr + ps;
      o[0][r] *= corr; o[1][r] *= corr; o[2][r] *= corr; o[3][r] *= corr;
    }
    // ---- re-fragment P: C-layout -> LDS row-major -> A-layout ----
#pragma unroll
    for (int r = 0; r < 8; ++r) {
      Plds[wave][r + 8*g][l16]      = f2bf(s0[r]);
      Plds[wave][r + 8*g][16 + l16] = f2bf(s1[r]);
    }
    v16bf pa = cat8(*(const v8bf*)&Plds[wave][l16][16*g],
                    *(const v8bf*)&Plds[wave][l16][16*g + 8]);
    // ---- drain async V copy, then O += P(16x32) x V(32x64) : 4 d-tiles ----
    asm volatile("s_wait_asynccnt 0x0" ::: "memory");
#pragma unroll
    for (int t = 0; t < 4; ++t) {
      const __bf16* vr = &Vlds[wave][16*t + l16][16*g];
      v16bf bv = cat8(*(const v8bf*)vr, *(const v8bf*)(vr + 8));
      o[t] = wmma_bf16(pa, bv, o[t]);
    }
  }
  // ---- normalize and store bf16 attn output [L][HQ*64] ----
#pragma unroll
  for (int r = 0; r < 8; ++r) {
    float inv = 1.f / lrow[r];
    __bf16* dst = Ob + (size_t)(q0 + r + 8*g) * HD + h * DH + l16;
    dst[0]  = f2bf(o[0][r] * inv);
    dst[16] = f2bf(o[1][r] * inv);
    dst[32] = f2bf(o[2][r] * inv);
    dst[48] = f2bf(o[3][r] * inv);
  }
}

// ---------- host ----------
extern "C" void kernel_launch(void* const* d_in, const int* in_sizes, int n_in,
                              void* d_out, int out_size, void* d_ws, size_t ws_size,
                              hipStream_t stream) {
  const float* x    = (const float*)d_in[0];
  const float* cosb = (const float*)d_in[1];
  const float* sinb = (const float*)d_in[2];
  const float* Wq   = (const float*)d_in[3];
  const float* Wk   = (const float*)d_in[4];
  const float* Wv   = (const float*)d_in[5];
  const float* Wo   = (const float*)d_in[6];
  float* out = (float*)d_out;

  char* p = (char*)d_ws;
  auto alloc = [&](size_t bytes) -> char* {
    char* r = p; p += (bytes + 255) & ~(size_t)255; return r;
  };
  __bf16* xbf  = (__bf16*)alloc((size_t)LSEQ * CIN * 2);
  __bf16* wqbf = (__bf16*)alloc((size_t)HD   * CIN * 2);
  __bf16* wkbf = (__bf16*)alloc((size_t)KVD  * CIN * 2);
  __bf16* wvbf = (__bf16*)alloc((size_t)KVD  * CIN * 2);
  __bf16* wobf = (__bf16*)alloc((size_t)HD   * HD  * 2);
  float*  qf   = (float*) alloc((size_t)LSEQ * HD  * 4);
  float*  kf   = (float*) alloc((size_t)LSEQ * KVD * 4);
  float*  vf   = (float*) alloc((size_t)LSEQ * KVD * 4);
  __bf16* qr   = (__bf16*)alloc((size_t)HQ  * LSEQ * DH * 2);
  __bf16* kr   = (__bf16*)alloc((size_t)HKV * LSEQ * DH * 2);
  __bf16* vt   = (__bf16*)alloc((size_t)HKV * DH * LSEQ * 2);
  __bf16* abf  = (__bf16*)alloc((size_t)LSEQ * HD * 2);

  auto blks = [](size_t n) { return dim3((unsigned)((n + 255) / 256)); };

  // 1) casts
  k_cast_bf16<<<blks((size_t)LSEQ*CIN), 256, 0, stream>>>(x,  xbf,  LSEQ*CIN);
  k_cast_bf16<<<blks((size_t)HD*CIN),   256, 0, stream>>>(Wq, wqbf, HD*CIN);
  k_cast_bf16<<<blks((size_t)KVD*CIN),  256, 0, stream>>>(Wk, wkbf, KVD*CIN);
  k_cast_bf16<<<blks((size_t)KVD*CIN),  256, 0, stream>>>(Wv, wvbf, KVD*CIN);
  k_cast_bf16<<<blks((size_t)HD*HD),    256, 0, stream>>>(Wo, wobf, HD*HD);

  // 2) QKV projections (WMMA GEMMs, fp32 accum out)
  int tiles;
  tiles = (LSEQ/16)*(HD/64);
  k_gemm_bf16_nt<<<(tiles+7)/8, 256, 0, stream>>>(xbf, wqbf, qf, LSEQ, HD,  CIN);
  tiles = (LSEQ/16)*(KVD/64);
  k_gemm_bf16_nt<<<(tiles+7)/8, 256, 0, stream>>>(xbf, wkbf, kf, LSEQ, KVD, CIN);
  k_gemm_bf16_nt<<<(tiles+7)/8, 256, 0, stream>>>(xbf, wvbf, vf, LSEQ, KVD, CIN);

  // 3) RoPE + head-major bf16 packs
  k_rope_pack<<<blks((size_t)LSEQ*HQ*DH),  256, 0, stream>>>(qf, cosb, sinb, qr, LSEQ, HQ);
  k_rope_pack<<<blks((size_t)LSEQ*HKV*DH), 256, 0, stream>>>(kf, cosb, sinb, kr, LSEQ, HKV);
  k_vpack_t  <<<blks((size_t)LSEQ*KVD),    256, 0, stream>>>(vf, vt, LSEQ);

  // 4) causal flash attention (WMMA + async V staging)
  k_flash_attn<<<(HQ*(LSEQ/16))/8, 256, 0, stream>>>(qr, kr, vt, abf, LSEQ);

  // 5) output projection
  tiles = (LSEQ/16)*(HD/64);
  k_gemm_bf16_nt<<<(tiles+7)/8, 256, 0, stream>>>(abf, wobf, out, LSEQ, HD, HD);
}
